// MAC_Cell_7679401525563
// MI455X (gfx1250) — matrix operand order repaired and checked by reading, not compiled, not run on hardware
//
#include <hip/hip_runtime.h>
#include <hip/hip_bf16.h>
#include <cstddef>

// ---------------------------------------------------------------------------
// Types for CDNA5 WMMA (wave32, 16x16x32 bf16 -> f32 accumulate)
// ---------------------------------------------------------------------------
typedef __attribute__((ext_vector_type(16))) __bf16 v16bf;
typedef __attribute__((ext_vector_type(8)))  float  v8f;

#if defined(__has_builtin)
#if __has_builtin(__builtin_amdgcn_global_load_async_to_lds_b32) && \
    __has_builtin(__builtin_amdgcn_s_wait_asynccnt)
#define HAVE_ASYNC_LDS 1
#endif
#endif
#ifndef HAVE_ASYNC_LDS
#define HAVE_ASYNC_LDS 0
#endif

// Native fptrunc (round-to-nearest-even) -> lets the backend use the
// hardware bf16 convert ops instead of a 3-VALU bit-twiddle per element.
static __device__ __forceinline__ __bf16 f2bf(float f) {
  return (__bf16)f;
}

static __device__ __forceinline__ float warp_sum(float v) {
#pragma unroll
  for (int off = 16; off; off >>= 1) v += __shfl_xor(v, off, 32);
  return v;
}

// ---------------------------------------------------------------------------
// Generic small GEMM: Out(M,N) = A(M,K) @ W (+ bias) (+ Cin), bf16 WMMA.
// TRANSW: B[k,n] = W[n*ldw + k]  (for the algebraic-collapse projections
// through W_rd^T / W_rk^T).  One wave per 16x16 tile; K multiple of 64.
// Non-transposed B tiles are cooperatively staged through LDS so global
// loads stay coalesced (no strided scalar-load chains).
// ---------------------------------------------------------------------------
template <bool TRANSW, bool HAS_BIAS, bool HAS_CIN>
__global__ __launch_bounds__(32)
void k_gemm_bf16(const float* __restrict__ A, int lda, int K,
                 const float* __restrict__ W, int ldw,
                 const float* __restrict__ bias,
                 const float* __restrict__ Cin, int ldc,
                 float* __restrict__ Out, int ldo) {
  const int lane = threadIdx.x & 31;
  const int lr = lane & 15, hi = lane >> 4;
  const int row0 = blockIdx.y * 16, col0 = blockIdx.x * 16;

  v8f c;
#pragma unroll
  for (int r = 0; r < 8; ++r) {
    float v = 0.f;
    if (HAS_CIN)  v  = Cin[(size_t)(row0 + r + 8 * hi) * ldc + col0 + lr];
    if (HAS_BIAS) v += bias[col0 + lr];
    c[r] = v;
  }

  // A 16x32 bf16 fragment: lane row = lr; lanes 0-15 K 0..15,
  // lanes 16-31 K 16..31.
  const float* Arow = A + (size_t)(row0 + lr) * lda + hi * 16;
  // B 32x16 bf16 fragment: lane col = lr; lanes 0-15 K 0..15, lanes 16-31 K 16..31.
  const float* Bt = W + (size_t)(col0 + lr) * ldw + hi * 16;   // TRANSW path

  __shared__ float bT[32][20];   // padded: float4-alignable rows, used if !TRANSW

  for (int kk = 0; kk < K; kk += 64) {
#pragma unroll
    for (int half = 0; half < 2; ++half) {
      const int k0 = kk + half * 32;
      v16bf a, bm;
      const float4* ap = (const float4*)(Arow + k0);
      float4 a0 = ap[0], a1 = ap[1], a2 = ap[2], a3 = ap[3];
      a[0]  = f2bf(a0.x); a[1]  = f2bf(a0.y); a[2]  = f2bf(a0.z); a[3]  = f2bf(a0.w);
      a[4]  = f2bf(a1.x); a[5]  = f2bf(a1.y); a[6]  = f2bf(a1.z); a[7]  = f2bf(a1.w);
      a[8]  = f2bf(a2.x); a[9]  = f2bf(a2.y); a[10] = f2bf(a2.z); a[11] = f2bf(a2.w);
      a[12] = f2bf(a3.x); a[13] = f2bf(a3.y); a[14] = f2bf(a3.z); a[15] = f2bf(a3.w);
      if (TRANSW) {
        const float4* bp = (const float4*)(Bt + k0);
        float4 b0 = bp[0], b1 = bp[1], b2 = bp[2], b3 = bp[3];
        bm[0]  = f2bf(b0.x); bm[1]  = f2bf(b0.y); bm[2]  = f2bf(b0.z); bm[3]  = f2bf(b0.w);
        bm[4]  = f2bf(b1.x); bm[5]  = f2bf(b1.y); bm[6]  = f2bf(b1.z); bm[7]  = f2bf(b1.w);
        bm[8]  = f2bf(b2.x); bm[9]  = f2bf(b2.y); bm[10] = f2bf(b2.z); bm[11] = f2bf(b2.w);
        bm[12] = f2bf(b3.x); bm[13] = f2bf(b3.y); bm[14] = f2bf(b3.z); bm[15] = f2bf(b3.w);
      } else {
        // Cooperative coalesced load of the 32(K) x 16(N) tile, LDS transpose.
        const float4* wr = (const float4*)(W + (size_t)(k0 + lane) * ldw + col0);
        float4 w0 = wr[0], w1 = wr[1], w2 = wr[2], w3 = wr[3];
        __syncthreads();                       // previous reads done
        float4* br = (float4*)&bT[lane][0];    // row stride 80B -> 16B aligned
        br[0] = w0; br[1] = w1; br[2] = w2; br[3] = w3;
        __syncthreads();                       // stores visible
#pragma unroll
        for (int e = 0; e < 16; ++e)
          bm[e] = f2bf(bT[e + hi * 16][lr]);
      }
      c = __builtin_amdgcn_wmma_f32_16x16x32_bf16(false, a, false, bm,
                                                  (short)0, c, false, false);
    }
  }

#pragma unroll
  for (int r = 0; r < 8; ++r)
    Out[(size_t)(row0 + r + 8 * hi) * ldo + col0 + lr] = c[r];
}

// ---------------------------------------------------------------------------
// Small elementwise / packing kernels
// ---------------------------------------------------------------------------
__global__ void k_pack_xe(const float* x, const float* qrep, float* xe, int n) {
  int i = blockIdx.x * blockDim.x + threadIdx.x;
  if (i < n) xe[i] = x[i] * qrep[i];
}
__global__ void k_pack_qspc(const float* qs, const float* h, float* dst) {
  int i = blockIdx.x * blockDim.x + threadIdx.x;
  if (i >= 64 * 1024) return;
  int b = i >> 10, d = i & 1023;
  dst[i] = (d < 512) ? qs[b * 512 + d] : h[(size_t)b * 12288 + (d - 512)];
}
__global__ void k_pack_pm(const float* h, float* pm) {
  int i = blockIdx.x * blockDim.x + threadIdx.x;
  if (i >= 64 * 512) return;
  int b = i >> 9, d = i & 511;
  pm[i] = h[(size_t)b * 12288 + 512 + d];
}
__global__ void k_pack_pmrd(const float* h, const float* rd, float* dst) {
  int i = blockIdx.x * blockDim.x + threadIdx.x;
  if (i >= 64 * 1024) return;
  int b = i >> 10, d = i & 1023;
  dst[i] = (d < 512) ? h[(size_t)b * 12288 + 512 + d] : rd[b * 512 + (d - 512)];
}

// ---------------------------------------------------------------------------
// Control unit: focus logits over LQ=64, softmax, new_control, u = nc * w_ret
// ---------------------------------------------------------------------------
__global__ __launch_bounds__(256)
void k_ctrl(const float* __restrict__ cq, const float* __restrict__ question,
            const float* __restrict__ w_f, const float* __restrict__ b_f,
            const float* __restrict__ w_ret, float* nc, float* u) {
  int b = blockIdx.x, tid = threadIdx.x, lane = tid & 31, wave = tid >> 5;
  __shared__ float cqw[512];
  __shared__ float foc[64];
  __shared__ float sinv;
  cqw[tid]       = cq[b * 512 + tid]       * w_f[tid];
  cqw[tid + 256] = cq[b * 512 + tid + 256] * w_f[tid + 256];
  __syncthreads();
  const float* qb = question + (size_t)b * 64 * 512;
  for (int i = 0; i < 8; ++i) {
    int l = wave * 8 + i;
    const float* qr = qb + (size_t)l * 512;
    float acc = 0.f;
#pragma unroll
    for (int j = 0; j < 16; ++j) acc += qr[lane + 32 * j] * cqw[lane + 32 * j];
    acc = warp_sum(acc);
    if (lane == 0) foc[l] = acc + b_f[0];
  }
  __syncthreads();
  if (tid == 0) {
    float m = foc[0];
    for (int l = 1; l < 64; ++l) m = fmaxf(m, foc[l]);
    float s = 0.f;
    for (int l = 0; l < 64; ++l) { float e = __expf(foc[l] - m); foc[l] = e; s += e; }
    sinv = 1.0f / (s * 64.0f);                 // softmax + mean over LQ
  }
  __syncthreads();
  float inv = sinv;
  for (int d = tid; d < 512; d += 256) {
    float acc = 0.f;
    for (int l = 0; l < 64; ++l) acc += foc[l] * qb[(size_t)l * 512 + d];
    float v = acc * inv;
    nc[b * 512 + d] = v;
    u[b * 512 + d]  = v * w_ret[d];
  }
}

// ---------------------------------------------------------------------------
// hvec = rm * c  (c = a_full[:,512:]), Cconst = (rm*b_rk)·c + b_rd·u + b_ret
// ---------------------------------------------------------------------------
__global__ __launch_bounds__(256)
void k_prep_g(const float* rm, const float* a_full, const float* b_rk,
              const float* u, const float* b_rd, const float* b_ret,
              float* hvec, float* Cconst) {
  int b = blockIdx.x, tid = threadIdx.x, lane = tid & 31, wave = tid >> 5;
  __shared__ float red[8];
  float acc = 0.f;
  for (int i = tid; i < 512; i += 256) {
    float cvi = a_full[(size_t)b * 1024 + 512 + i];
    float rmi = rm[b * 512 + i];
    hvec[b * 512 + i] = rmi * cvi;
    acc += rmi * b_rk[i] * cvi + b_rd[i] * u[b * 512 + i];
  }
  acc = warp_sum(acc);
  if (lane == 0) red[wave] = acc;
  __syncthreads();
  if (tid == 0) {
    float s = 0.f;
    for (int w = 0; w < 8; ++w) s += red[w];
    Cconst[b] = s + b_ret[0];
  }
}

// ---------------------------------------------------------------------------
// The bandwidth kernel: one pass over knowledge (256 MB). Per (b, chunk of
// 128 rows): logits z = kn·g + C, chunk max m, s = sum(exp(z-m)), and
// v[d] = sum_l exp(z_l-m) * kn[l,d]  (chunk re-read hits L2: 256 KB << 192 MB).
// g vector staged to LDS via CDNA5 async global->LDS path when available.
// ---------------------------------------------------------------------------
__global__ __launch_bounds__(256)
void k_main(const float* __restrict__ kn, const float* __restrict__ g,
            const float* __restrict__ Cc, float* part_ms, float* part_v) {
  int chunk = blockIdx.x, b = blockIdx.y;
  int tid = threadIdx.x, lane = tid & 31, wave = tid >> 5;
  __shared__ float gs[512];
  __shared__ float zb[128];
  __shared__ float red[8];
  __shared__ float s_m;
  const float* gb = g + b * 512;
#if HAVE_ASYNC_LDS
  {
    typedef __attribute__((address_space(1))) int gint_t;
    typedef __attribute__((address_space(3))) int lint_t;
    __builtin_amdgcn_global_load_async_to_lds_b32(
        (gint_t*)(gb + tid), (lint_t*)&gs[tid], 0, 0);
    __builtin_amdgcn_global_load_async_to_lds_b32(
        (gint_t*)(gb + tid + 256), (lint_t*)&gs[tid + 256], 0, 0);
    __builtin_amdgcn_s_wait_asynccnt(0);
  }
#else
  gs[tid]       = gb[tid];
  gs[tid + 256] = gb[tid + 256];
#endif
  __syncthreads();
  const float* knb = kn + ((size_t)b * 2048 + (size_t)chunk * 128) * 512;
  float Cb = Cc[b];
  for (int r = 0; r < 16; ++r) {
    int row = wave * 16 + r;
    const float4* kr = (const float4*)(knb + (size_t)row * 512);
    if (r + 1 < 16)
      __builtin_prefetch(knb + (size_t)(row + 1) * 512 + lane * 16, 0, 0);
    float acc = 0.f;
#pragma unroll
    for (int j = 0; j < 4; ++j) {
      float4 v = kr[lane + 32 * j];
      int base = (lane + 32 * j) * 4;
      acc += v.x * gs[base] + v.y * gs[base + 1] + v.z * gs[base + 2] + v.w * gs[base + 3];
    }
    acc = warp_sum(acc);
    if (lane == 0) zb[row] = acc + Cb;
  }
  __syncthreads();
  // chunk max
  float m = (tid < 128) ? zb[tid] : -3.4e38f;
#pragma unroll
  for (int off = 16; off; off >>= 1) m = fmaxf(m, __shfl_xor(m, off, 32));
  if (lane == 0) red[wave] = m;
  __syncthreads();
  if (tid == 0) {
    float mm = red[0];
    for (int w = 1; w < 8; ++w) mm = fmaxf(mm, red[w]);
    s_m = mm;
  }
  __syncthreads();
  float mx = s_m;
  float e = 0.f;
  if (tid < 128) { e = __expf(zb[tid] - mx); zb[tid] = e; }
  float s = warp_sum(e);
  if (lane == 0) red[wave] = s;
  __syncthreads();
  if (tid == 0) {
    float ss = 0.f;
    for (int w = 0; w < 8; ++w) ss += red[w];
    part_ms[(b * 16 + chunk) * 2 + 0] = mx;
    part_ms[(b * 16 + chunk) * 2 + 1] = ss;
  }
  __syncthreads();
  // exp-weighted accumulation of knowledge rows (chunk is L2-resident)
  float a0 = 0.f, a1 = 0.f;
  int d0 = tid, d1 = tid + 256;
  for (int l = 0; l < 128; ++l) {
    float el = zb[l];
    const float* kr = knb + (size_t)l * 512;
    a0 += el * kr[d0];
    a1 += el * kr[d1];
  }
  float* pv = part_v + ((size_t)(b * 16 + chunk)) * 512;
  pv[d0] = a0;
  pv[d1] = a1;
}

// Combine flash-style partials: read[b,d] = sum_c e^{m_c-M} v_c[d] / (S*LK)
__global__ __launch_bounds__(256)
void k_combine(const float* part_ms, const float* part_v, float* readv) {
  int b = blockIdx.x, tid = threadIdx.x;
  __shared__ float sc[16];
  __shared__ float sinv;
  if (tid == 0) {
    float M = -3.4e38f;
    for (int c = 0; c < 16; ++c) M = fmaxf(M, part_ms[(b * 16 + c) * 2]);
    float S = 0.f;
    for (int c = 0; c < 16; ++c) {
      float e = __expf(part_ms[(b * 16 + c) * 2] - M);
      sc[c] = e;
      S += e * part_ms[(b * 16 + c) * 2 + 1];
    }
    sinv = 1.0f / (S * 2048.0f);
  }
  __syncthreads();
  float inv = sinv;
  for (int d = tid; d < 512; d += 256) {
    float acc = 0.f;
    for (int c = 0; c < 16; ++c) acc += sc[c] * part_v[((size_t)(b * 16 + c)) * 512 + d];
    readv[b * 512 + d] = acc * inv;
  }
}

// ---------------------------------------------------------------------------
// Write unit: ctrl-attention over S=12 past states (with ca==0 mask), msa,
// gate = sigmoid(nc·w_m3 + b_m3)
// ---------------------------------------------------------------------------
__global__ __launch_bounds__(128)
void k_wattn(const float* __restrict__ h, const float* __restrict__ nc,
             const float* w_ca, const float* b_ca,
             const float* w_m3, const float* b_m3,
             float* msa, float* gate) {
  int b = blockIdx.x, tid = threadIdx.x, lane = tid & 31, wave = tid >> 5;
  __shared__ float ncw[512];
  __shared__ float cas[12];
  __shared__ float watt[12];
  __shared__ float red[4];
  float gacc = 0.f;
  for (int i = tid; i < 512; i += 128) {
    float n = nc[b * 512 + i];
    ncw[i] = n * w_ca[i];
    gacc += n * w_m3[i];
  }
  gacc = warp_sum(gacc);
  if (lane == 0) red[wave] = gacc;
  __syncthreads();
  if (tid == 0) {
    float sdot = red[0] + red[1] + red[2] + red[3] + b_m3[0];
    gate[b] = 1.0f / (1.0f + __expf(-sdot));
  }
  const float* hb = h + (size_t)b * 12288;
  for (int s = wave; s < 12; s += 4) {
    float acc = 0.f;
#pragma unroll
    for (int j = 0; j < 16; ++j)
      acc += hb[(size_t)s * 1024 + lane + 32 * j] * ncw[lane + 32 * j];
    acc = warp_sum(acc);
    if (lane == 0) cas[s] = acc + b_ca[0];
  }
  __syncthreads();
  if (tid == 0) {
    float m = -3.4e38f;
    for (int s = 0; s < 12; ++s) {
      float c = cas[s];
      if (c == 0.0f) c = -1e9f;                // pad mask from reference
      cas[s] = c;
      m = fmaxf(m, c);
    }
    float S = 0.f;
    for (int s = 0; s < 12; ++s) { float e = __expf(cas[s] - m); watt[s] = e; S += e; }
    for (int s = 0; s < 12; ++s) watt[s] /= S;
  }
  __syncthreads();
  for (int d = tid; d < 512; d += 128) {
    float acc = 0.f;
    for (int s = 0; s < 12; ++s) acc += watt[s] * hb[(size_t)s * 1024 + 512 + d];
    msa[b * 512 + d] = acc;
  }
}

// new_state row: [nc | mp*g + (1-g)*pm]
__global__ __launch_bounds__(256)
void k_final(const float* nc, const float* mp, const float* gate,
             const float* h, float* out) {
  int b = blockIdx.x, tid = threadIdx.x;
  float gt = gate[b];
  for (int d = tid; d < 1024; d += 256) {
    float v;
    if (d < 512) v = nc[b * 512 + d];
    else {
      float pm = h[(size_t)b * 12288 + 512 + (d - 512)];
      v = mp[b * 512 + (d - 512)] * gt + (1.0f - gt) * pm;
    }
    out[(size_t)b * 12288 + d] = v;
  }
}

// out[:,1:12,:] = h[:,0:11,:]
__global__ void k_shift(const float* h, float* out) {
  size_t idx = (size_t)blockIdx.x * blockDim.x + threadIdx.x;
  const size_t total = 64ULL * 11 * 1024;
  if (idx < total) {
    size_t b = idx / (11 * 1024), o = idx % (11 * 1024);
    out[b * 12288 + 1024 + o] = h[b * 12288 + o];
  }
}

// ---------------------------------------------------------------------------
// Host launcher
// ---------------------------------------------------------------------------
extern "C" void kernel_launch(void* const* d_in, const int* in_sizes, int n_in,
                              void* d_out, int out_size, void* d_ws, size_t ws_size,
                              hipStream_t stream) {
  // setup_inputs order, params pytree flattened by sorted keys (b before w):
  const float* x     = (const float*)d_in[0];   // (64,512)
  const float* h     = (const float*)d_in[1];   // (64,12,1024)
  const float* kn    = (const float*)d_in[2];   // (64,2048,512)
  const float* quest = (const float*)d_in[3];   // (64,64,512)
  const float* qrep  = (const float*)d_in[4];   // (64,512)
  const float* b_cq  = (const float*)d_in[5];
  const float* W_cq  = (const float*)d_in[6];   // (1024,512)
  const float* b_f   = (const float*)d_in[7];
  const float* w_f   = (const float*)d_in[8];   // (512,)
  const float* b_qs  = (const float*)d_in[9];
  const float* W_qs  = (const float*)d_in[10];  // (512,512)
  const float* b_rd  = (const float*)d_in[11];
  const float* W_rd  = (const float*)d_in[12];  // (1024,512)
  const float* b_rk  = (const float*)d_in[13];
  const float* W_rk  = (const float*)d_in[14];  // (512,512)
  const float* b_rm  = (const float*)d_in[15];
  const float* W_rm  = (const float*)d_in[16];  // (512,512)
  const float* b_ret = (const float*)d_in[17];
  const float* w_ret = (const float*)d_in[18];  // (512,)
  const float* b_ca  = (const float*)d_in[19];
  const float* w_ca  = (const float*)d_in[20];  // (512,)
  const float* b_m1  = (const float*)d_in[21];
  const float* W_m1  = (const float*)d_in[22];  // (1024,512)
  const float* b_m2  = (const float*)d_in[23];
  const float* W_m2  = (const float*)d_in[24];  // (512,512)
  const float* b_m3  = (const float*)d_in[25];
  const float* w_m3  = (const float*)d_in[26];  // (512,)
  const float* b_s   = (const float*)d_in[27];
  const float* W_s   = (const float*)d_in[28];  // (512,512)
  float* out = (float*)d_out;

  float* W = (float*)d_ws;
  float* XE   = W;            // 32768
  float* QS   = W + 32768;    // 32768
  float* QSPC = W + 65536;    // 65536
  float* CQ   = W + 131072;   // 32768
  float* NC   = W + 163840;   // 32768
  float* U    = W + 196608;   // 32768
  float* AF   = W + 229376;   // 65536  (a | c) per batch
  float* PM   = W + 294912;   // 32768
  float* RM   = W + 327680;   // 32768
  float* HV   = W + 360448;   // 32768
  float* G    = W + 393216;   // 32768
  float* CC   = W + 425984;   // 64
  float* PMS  = W + 426048;   // 2048
  float* PV   = W + 428096;   // 524288
  float* RD   = W + 952384;   // 32768
  float* PMRD = W + 985152;   // 65536
  float* M1   = W + 1050688;  // 32768
  float* MSA  = W + 1083456;  // 32768
  float* GATE = W + 1116224;  // 64
  float* MP   = W + 1116288;  // 32768

  dim3 g32x4(32, 4), g64x4(64, 4);

  // 1) quest_state = (x*qrep) @ W_qs + b
  k_pack_xe<<<128, 256, 0, stream>>>(x, qrep, XE, 64 * 512);
  k_gemm_bf16<false, true, false><<<g32x4, 32, 0, stream>>>(
      XE, 512, 512, W_qs, 512, b_qs, nullptr, 0, QS, 512);
  // 2) cq = [qs, prev_control] @ W_cq + b
  k_pack_qspc<<<256, 256, 0, stream>>>(QS, h, QSPC);
  k_gemm_bf16<false, true, false><<<g32x4, 32, 0, stream>>>(
      QSPC, 1024, 1024, W_cq, 512, b_cq, nullptr, 0, CQ, 512);
  // 3) control attention -> nc, u = nc*w_ret
  k_ctrl<<<64, 256, 0, stream>>>(CQ, quest, w_f, b_f, w_ret, NC, U);
  // 4) [a|c] = u @ W_rd^T  (rank-1 collapse of rd_disjoint)
  k_gemm_bf16<true, false, false><<<g64x4, 32, 0, stream>>>(
      U, 512, 512, W_rd, 512, nullptr, nullptr, 0, AF, 1024);
  // 5) rm = pm @ W_rm + b
  k_pack_pm<<<128, 256, 0, stream>>>(h, PM);
  k_gemm_bf16<false, true, false><<<g32x4, 32, 0, stream>>>(
      PM, 512, 512, W_rm, 512, b_rm, nullptr, 0, RM, 512);
  // 6) hvec = rm*c ; Cconst
  k_prep_g<<<64, 256, 0, stream>>>(RM, AF, b_rk, U, b_rd, b_ret, HV, CC);
  // 7) g = a + hvec @ W_rk^T   (rank-1 collapse of rd_knowledge)
  k_gemm_bf16<true, false, true><<<g32x4, 32, 0, stream>>>(
      HV, 512, 512, W_rk, 512, nullptr, AF, 1024, G, 512);
  // 8) single streaming pass over knowledge (flash-style partials)
  k_main<<<dim3(16, 64), 256, 0, stream>>>(kn, G, CC, PMS, PV);
  k_combine<<<64, 256, 0, stream>>>(PMS, PV, RD);
  // 9) m1 = [pm, read] @ W_m1 + b
  k_pack_pmrd<<<256, 256, 0, stream>>>(h, RD, PMRD);
  k_gemm_bf16<false, true, false><<<g32x4, 32, 0, stream>>>(
      PMRD, 1024, 1024, W_m1, 512, b_m1, nullptr, 0, M1, 512);
  // 10) write-attention over past states + gate
  k_wattn<<<64, 128, 0, stream>>>(h, NC, w_ca, b_ca, w_m3, b_m3, MSA, GATE);
  // 11) mp = m1 @ W_m2 + b_m2 ; mp += msa @ W_s + b_s
  k_gemm_bf16<false, true, false><<<g32x4, 32, 0, stream>>>(
      M1, 512, 512, W_m2, 512, b_m2, nullptr, 0, MP, 512);
  k_gemm_bf16<false, true, true><<<g32x4, 32, 0, stream>>>(
      MSA, 512, 512, W_s, 512, b_s, MP, 512, MP, 512);
  // 12) blend + emit h_new
  k_final<<<64, 256, 0, stream>>>(NC, MP, GATE, h, out);
  k_shift<<<2816, 256, 0, stream>>>(h, out);
}
